// SnippetShotQueryGCN_31430570672681
// MI455X (gfx1250) — matrix operand order repaired and checked by reading, not compiled, not run on hardware
//
#include <hip/hip_runtime.h>

typedef __attribute__((ext_vector_type(16))) _Float16 v16h;
typedef __attribute__((ext_vector_type(8)))  _Float16 v8h;
typedef __attribute__((ext_vector_type(8)))  float    v8f;

namespace {
constexpr int Bn = 8, Cn = 256, Tn = 4096, Mn = 64, Kk = 6, Wn = 128;
constexpr float EPSn = 1e-5f;
}

// ---------------- WMMA fragment loaders (CDNA5 wave32 layouts) ----------------
// A (16x32 f16, row-major source): lanes 0-15 row M=lane, K = k0+{0..7, 16..23};
// lanes 16-31 row M=lane-16, K = k0+{8..15, 24..31}.
__device__ __forceinline__ v16h load_a_rm(const _Float16* __restrict__ A, int row,
                                          int ld, int k0, int lane) {
  const _Float16* p = A + (size_t)row * ld + k0 + ((lane >> 4) << 3);
  v8h lo = *(const v8h*)p;
  v8h hi = *(const v8h*)(p + 16);
  v16h r;
#pragma unroll
  for (int i = 0; i < 8; ++i) { r[i] = lo[i]; r[i + 8] = hi[i]; }
  return r;
}
// B (32x16 f16): lane = N column; lanes 0-15 hold K=k0+{0..7, 16..23}, lanes 16-31 K=k0+{8..15, 24..31}.
// Source stored point-major (row = point/column, K contiguous) -> two 16B loads.
__device__ __forceinline__ v16h load_b_pm(const _Float16* __restrict__ Bp, int col,
                                          int ld, int k0, int lane) {
  const _Float16* p = Bp + (size_t)col * ld + k0 + ((lane >> 4) << 3);
  v8h lo = *(const v8h*)p;
  v8h hi = *(const v8h*)(p + 16);
  v16h r;
#pragma unroll
  for (int i = 0; i < 8; ++i) { r[i] = lo[i]; r[i + 8] = hi[i]; }
  return r;
}

// ---------------- small utility kernels ----------------
__global__ void k_f32_to_f16(const float* __restrict__ s, _Float16* __restrict__ d, int n) {
  int i = blockIdx.x * 256 + threadIdx.x;
  if (i < n) d[i] = (_Float16)s[i];
}
// s1 weight (128 x 512): diff[w][c] = s1w[w][256+c] - s1w[w][c]  (ego-part minus neigh-part)
__global__ void k_s1diff(const float* __restrict__ s1w, _Float16* __restrict__ d) {
  int i = blockIdx.x * 256 + threadIdx.x;  // 128*256
  int w = i >> 8, c = i & 255;
  d[i] = (_Float16)(s1w[(size_t)w * 512 + 256 + c] - s1w[(size_t)w * 512 + c]);
}

// ---------------- backbone1: grouped conv3 (groups=4) + BN + ReLU ----------------
__global__ void k_backbone1(const float* __restrict__ x, const float* __restrict__ w,
                            const float* __restrict__ bias, const float* __restrict__ g,
                            const float* __restrict__ be, const float* __restrict__ mu,
                            const float* __restrict__ var, float* __restrict__ out) {
  int t = blockIdx.x * 256 + threadIdx.x;
  int o = blockIdx.y, b = blockIdx.z;
  int og = o >> 6;  // 64 in-ch per group
  float acc = bias[o];
  for (int ci = 0; ci < 64; ++ci) {
    int cin = og * 64 + ci;
    const float* xr = x + ((size_t)b * Cn + cin) * Tn;
    const float* wr = w + ((size_t)o * 64 + ci) * 3;
    float xm = (t > 0) ? xr[t - 1] : 0.f;
    float xp = (t < Tn - 1) ? xr[t + 1] : 0.f;
    acc += wr[0] * xm + wr[1] * xr[t] + wr[2] * xp;
  }
  float y = g[o] * (acc - mu[o]) * rsqrtf(var[o] + EPSn) + be[o];
  out[((size_t)b * Cn + o) * Tn + t] = fmaxf(y, 0.f);
}

// ---------------- backbone_topic: grouped conv1 (16->256, groups=4) + BN + ReLU ----
__global__ void k_backbone_topic(const float* __restrict__ topic, const float* __restrict__ w,
                                 const float* __restrict__ bias, const float* __restrict__ g,
                                 const float* __restrict__ be, const float* __restrict__ mu,
                                 const float* __restrict__ var, float* __restrict__ tf) {
  int m = threadIdx.x;          // 64
  int o = blockIdx.y, b = blockIdx.z;
  int og = o >> 6;              // 4 in-ch per group
  float acc = bias[o];
  const float* tp = topic + ((size_t)b * Mn + m) * 16;
#pragma unroll
  for (int ci = 0; ci < 4; ++ci) acc += w[o * 4 + ci] * tp[og * 4 + ci];
  float y = g[o] * (acc - mu[o]) * rsqrtf(var[o] + EPSn) + be[o];
  tf[((size_t)b * Cn + o) * Mn + m] = fmaxf(y, 0.f);
}

// ---------------- pack x (B,C,T) f32 -> point-major (B,T,C) f16 via LDS transpose ---
__global__ void k_pack_x(const float* __restrict__ x, _Float16* __restrict__ xh) {
  __shared__ float tile[32][33];
  int b = blockIdx.z;
  int n0 = blockIdx.x * 32, c0 = blockIdx.y * 32;
#pragma unroll
  for (int i = 0; i < 4; ++i) {
    int c = c0 + threadIdx.y + i * 8;
    tile[threadIdx.y + i * 8][threadIdx.x] = x[((size_t)b * Cn + c) * Tn + n0 + threadIdx.x];
  }
  __syncthreads();
#pragma unroll
  for (int i = 0; i < 4; ++i) {
    int n = n0 + threadIdx.y + i * 8;
    xh[((size_t)b * Tn + n) * Cn + c0 + threadIdx.x] = (_Float16)tile[threadIdx.x][threadIdx.y + i * 8];
  }
}

// ---------------- WMMA GEMM: t1 = relu(t1w @ x + b), channel-major f32 out ---------
__global__ void k_gemm_t1(const _Float16* __restrict__ A, const _Float16* __restrict__ xh,
                          const float* __restrict__ bias, float* __restrict__ out) {
  int lane = threadIdx.x;
  int o0 = blockIdx.x * 16, n0 = blockIdx.y * 16, b = blockIdx.z;
  const _Float16* Bp = xh + (size_t)b * Tn * Cn;
  v8f acc = {};
  for (int k0 = 0; k0 < Cn; k0 += 32) {
    v16h a = load_a_rm(A, o0 + (lane & 15), Cn, k0, lane);
    v16h bb = load_b_pm(Bp, n0 + (lane & 15), Cn, k0, lane);
    acc = __builtin_amdgcn_wmma_f32_16x16x32_f16(false, a, false, bb, (short)0, acc, false, false);
  }
  int n = n0 + (lane & 15), ob = o0 + ((lane >> 4) << 3);
#pragma unroll
  for (int r = 0; r < 8; ++r) {
    int o = ob + r;
    out[((size_t)b * Wn + o) * Tn + n] = fmaxf(acc[r] + bias[o], 0.f);
  }
}

// ---------------- WMMA GEMM: Q = (s1w_b - s1w_a) @ x, point-major f32 out ----------
__global__ void k_gemm_q(const _Float16* __restrict__ A, const _Float16* __restrict__ xh,
                         float* __restrict__ q) {
  int lane = threadIdx.x;
  int o0 = blockIdx.x * 16, n0 = blockIdx.y * 16, b = blockIdx.z;
  const _Float16* Bp = xh + (size_t)b * Tn * Cn;
  v8f acc = {};
  for (int k0 = 0; k0 < Cn; k0 += 32) {
    v16h a = load_a_rm(A, o0 + (lane & 15), Cn, k0, lane);
    v16h bb = load_b_pm(Bp, n0 + (lane & 15), Cn, k0, lane);
    acc = __builtin_amdgcn_wmma_f32_16x16x32_f16(false, a, false, bb, (short)0, acc, false, false);
  }
  int n = n0 + (lane & 15), ob = o0 + ((lane >> 4) << 3);
#pragma unroll
  for (int r = 0; r < 8; ++r) q[((size_t)b * Tn + n) * Wn + ob + r] = acc[r];
}

// ---------------- t2: grouped conv3 (groups=32) + ReLU -> point-major f16 ----------
__global__ void k_t2(const float* __restrict__ t1o, const float* __restrict__ w,
                     const float* __restrict__ bias, _Float16* __restrict__ t2h) {
  int t = blockIdx.x * 256 + threadIdx.x;
  int o = blockIdx.y, b = blockIdx.z;
  int g = o >> 2;
  float acc = bias[o];
#pragma unroll
  for (int ci = 0; ci < 4; ++ci) {
    const float* xr = t1o + ((size_t)b * Wn + g * 4 + ci) * Tn;
    const float* wr = w + ((size_t)o * 4 + ci) * 3;
    float xm = (t > 0) ? xr[t - 1] : 0.f;
    float xp = (t < Tn - 1) ? xr[t + 1] : 0.f;
    acc += wr[0] * xm + wr[1] * xr[t] + wr[2] * xp;
  }
  t2h[((size_t)b * Tn + t) * Wn + o] = (_Float16)fmaxf(acc, 0.f);
}

// ---------------- KNN: score = 2*<x,t_m> - ||t_m||^2 (||x||^2 is rank-invariant) ---
__global__ void k_dist_topk(const float* __restrict__ x, const float* __restrict__ tf,
                            int* __restrict__ idx) {
  __shared__ float sc[Mn];
  int m = threadIdx.x;
  int n = blockIdx.x, b = blockIdx.z;
  const float* xc = x + (size_t)b * Cn * Tn + n;
  const float* tc = tf + (size_t)b * Cn * Mn + m;
  float inner = 0.f, yy = 0.f;
  for (int c = 0; c < Cn; ++c) {
    float tv = tc[(size_t)c * Mn];
    yy += tv * tv;
    inner += xc[(size_t)c * Tn] * tv;
  }
  sc[m] = 2.f * inner - yy;
  __syncthreads();
  if (m == 0) {
    unsigned long long used = 0ull;
    int* ip = idx + ((size_t)b * Tn + n) * Kk;
    for (int j = 0; j < Kk; ++j) {
      float best = -3.4e38f; int bi = 0;
      for (int mm = 0; mm < Mn; ++mm)
        if (!((used >> mm) & 1ull) && sc[mm] > best) { best = sc[mm]; bi = mm; }
      used |= 1ull << bi;
      ip[j] = bi;
    }
  }
}

// ---------------- P[w,m] = s1w_a @ tf  (tiny, fp32 VALU) ----------------
__global__ void k_pmat(const float* __restrict__ s1w, const float* __restrict__ tf,
                       float* __restrict__ P) {
  int m = threadIdx.x;  // 64
  int w = blockIdx.y, b = blockIdx.z;
  float acc = 0.f;
  for (int c = 0; c < Cn; ++c)
    acc += s1w[(size_t)w * 512 + c] * tf[((size_t)b * Cn + c) * Mn + m];
  P[((size_t)b * Wn + w) * Mn + m] = acc;
}

// ---------------- fused s1 (gather P + Q + bias, relu) + s2 grouped 1x1 ------------
__global__ void k_s1s2(const float* __restrict__ P, const float* __restrict__ Q,
                       const int* __restrict__ idx, const float* __restrict__ s1b,
                       const float* __restrict__ s2w, const float* __restrict__ s2b,
                       _Float16* __restrict__ s2h) {
  __shared__ float sh[Wn];
  int w = threadIdx.x;  // 128
  int j = blockIdx.x, n = blockIdx.y, b = blockIdx.z;
  int m = idx[((size_t)b * Tn + n) * Kk + j];
  float s1 = P[((size_t)b * Wn + w) * Mn + m] + Q[((size_t)b * Tn + n) * Wn + w] + s1b[w];
  sh[w] = fmaxf(s1, 0.f);
  __syncthreads();
  int g = w >> 2;
  float acc = s2b[w];
#pragma unroll
  for (int ci = 0; ci < 4; ++ci) acc += s2w[w * 4 + ci] * sh[g * 4 + ci];
  s2h[(((size_t)b * Tn + n) * Kk + j) * Wn + w] = (_Float16)fmaxf(acc, 0.f);
}

// ---------------- block-final: t3 GEMM + 6x s3 GEMM + max_k + residual + ReLU ------
__global__ void k_final(const _Float16* __restrict__ t3wh, const _Float16* __restrict__ s3wh,
                        const _Float16* __restrict__ t2h, const _Float16* __restrict__ s2h,
                        const float* __restrict__ xin, const float* __restrict__ t3b,
                        const float* __restrict__ s3b, const float* __restrict__ identity,
                        float* __restrict__ out) {
  int lane = threadIdx.x;
  int o0 = blockIdx.x * 16, n0 = blockIdx.y * 16, b = blockIdx.z;
  const _Float16* Bt = t2h + (size_t)b * Tn * Wn;
  const _Float16* Bs = s2h + (size_t)b * Tn * Kk * Wn;
  int col = n0 + (lane & 15);
  v8f accT = {};
  v8f accS[Kk];
#pragma unroll
  for (int j = 0; j < Kk; ++j) accS[j] = accT;
  for (int k0 = 0; k0 < Wn; k0 += 32) {
    v16h aT = load_a_rm(t3wh, o0 + (lane & 15), Wn, k0, lane);
    v16h aS = load_a_rm(s3wh, o0 + (lane & 15), Wn, k0, lane);
    v16h bT = load_b_pm(Bt, col, Wn, k0, lane);
    if (k0 + 32 < Wn) __builtin_prefetch(Bt + (size_t)col * Wn + k0 + 32, 0, 1);
    accT = __builtin_amdgcn_wmma_f32_16x16x32_f16(false, aT, false, bT, (short)0, accT, false, false);
#pragma unroll
    for (int j = 0; j < Kk; ++j) {
      v16h bS = load_b_pm(Bs, col * Kk + j, Wn, k0, lane);
      accS[j] = __builtin_amdgcn_wmma_f32_16x16x32_f16(false, aS, false, bS, (short)0, accS[j], false, false);
    }
  }
  int n = col, ob = o0 + ((lane >> 4) << 3);
#pragma unroll
  for (int r = 0; r < 8; ++r) {
    int o = ob + r;
    float sm = accS[0][r];
#pragma unroll
    for (int j = 1; j < Kk; ++j) sm = fmaxf(sm, accS[j][r]);
    size_t oi = ((size_t)b * Cn + o) * Tn + n;
    float v = xin[oi] + accT[r] + t3b[o] + sm + s3b[o];
    v = fmaxf(v, 0.f);
    if (identity) v += identity[oi];
    out[oi] = v;
  }
}

// ---------------- host orchestration ----------------
struct GP {
  const float *t1w, *t1b, *t2w, *t2b, *t3w, *t3b, *s1w, *s1b, *s2w, *s2b, *s3w, *s3b;
};

extern "C" void kernel_launch(void* const* d_in, const int* in_sizes, int n_in,
                              void* d_out, int out_size, void* d_ws, size_t ws_size,
                              hipStream_t stream) {
  (void)n_in; (void)out_size; (void)ws_size;
  const float *snip, *topic;
  const float *bb1_w, *bb1_b, *bb1_g, *bb1_be, *bb1_m, *bb1_v;
  const float *bbt_w, *bbt_b, *bbt_g, *bbt_be, *bbt_m, *bbt_v;
  GP g[2];
  auto F = [&](int i) { return (const float*)d_in[i]; };
  if (in_sizes[0] == Bn * Cn * Tn) {
    // insertion-order flatten: snip, topic, bb1_w, bb1_b, bn(4), bbt_w, bbt_b, bn(4), g1.., g2..
    snip = F(0); topic = F(1);
    bb1_w = F(2); bb1_b = F(3); bb1_g = F(4); bb1_be = F(5); bb1_m = F(6); bb1_v = F(7);
    bbt_w = F(8); bbt_b = F(9); bbt_g = F(10); bbt_be = F(11); bbt_m = F(12); bbt_v = F(13);
    for (int i = 0; i < 2; ++i) {
      int o = 14 + i * 12;
      g[i] = {F(o + 0), F(o + 1), F(o + 2), F(o + 3), F(o + 4), F(o + 5),
              F(o + 6), F(o + 7), F(o + 8), F(o + 9), F(o + 10), F(o + 11)};
    }
  } else {
    // jax pytree flatten (dict keys sorted): params(bb1_b, bb1_bn, bb1_w, bbt_b, bbt_bn,
    // bbt_w, g1{s1b,s1w,s2b,s2w,s3b,s3w,t1b,t1w,t2b,t2w,t3b,t3w}, g2{...}), snip, topic
    bb1_b = F(0); bb1_g = F(1); bb1_be = F(2); bb1_m = F(3); bb1_v = F(4); bb1_w = F(5);
    bbt_b = F(6); bbt_g = F(7); bbt_be = F(8); bbt_m = F(9); bbt_v = F(10); bbt_w = F(11);
    for (int i = 0; i < 2; ++i) {
      int o = 12 + i * 12;
      g[i].s1b = F(o + 0); g[i].s1w = F(o + 1); g[i].s2b = F(o + 2); g[i].s2w = F(o + 3);
      g[i].s3b = F(o + 4); g[i].s3w = F(o + 5); g[i].t1b = F(o + 6); g[i].t1w = F(o + 7);
      g[i].t2b = F(o + 8); g[i].t2w = F(o + 9); g[i].t3b = F(o + 10); g[i].t3w = F(o + 11);
    }
    snip = F(36); topic = F(37);
  }

  // ---- workspace carve (256B aligned) ----
  size_t off = 0;
  auto carve = [&](size_t bytes) {
    void* p = (char*)d_ws + off;
    off += (bytes + 255) & ~(size_t)255;
    return p;
  };
  _Float16* xh    = (_Float16*)carve((size_t)Bn * Tn * Cn * 2);
  float*    base0 = (float*)   carve((size_t)Bn * Cn * Tn * 4);
  float*    base1 = (float*)   carve((size_t)Bn * Cn * Tn * 4);
  float*    tf    = (float*)   carve((size_t)Bn * Cn * Mn * 4);
  float*    t1o   = (float*)   carve((size_t)Bn * Wn * Tn * 4);
  _Float16* t2h   = (_Float16*)carve((size_t)Bn * Tn * Wn * 2);
  float*    qbuf  = (float*)   carve((size_t)Bn * Tn * Wn * 4);
  float*    pbuf  = (float*)   carve((size_t)Bn * Wn * Mn * 4);
  int*      idxb  = (int*)     carve((size_t)Bn * Tn * Kk * 4);
  _Float16* s2h   = (_Float16*)carve((size_t)Bn * Tn * Kk * Wn * 2);
  _Float16* t1wh[2], *t3wh[2], *s1dh[2], *s3wh[2];
  for (int i = 0; i < 2; ++i) {
    t1wh[i] = (_Float16*)carve((size_t)Wn * Cn * 2);
    t3wh[i] = (_Float16*)carve((size_t)Cn * Wn * 2);
    s1dh[i] = (_Float16*)carve((size_t)Wn * Cn * 2);
    s3wh[i] = (_Float16*)carve((size_t)Cn * Wn * 2);
  }

  // ---- weight precision conversion ----
  for (int i = 0; i < 2; ++i) {
    k_f32_to_f16<<<Wn * Cn / 256, 256, 0, stream>>>(g[i].t1w, t1wh[i], Wn * Cn);
    k_f32_to_f16<<<Cn * Wn / 256, 256, 0, stream>>>(g[i].t3w, t3wh[i], Cn * Wn);
    k_f32_to_f16<<<Cn * Wn / 256, 256, 0, stream>>>(g[i].s3w, s3wh[i], Cn * Wn);
    k_s1diff   <<<Wn * Cn / 256, 256, 0, stream>>>(g[i].s1w, s1dh[i]);
  }

  // ---- backbones ----
  k_backbone1<<<dim3(Tn / 256, Cn, Bn), 256, 0, stream>>>(
      snip, bb1_w, bb1_b, bb1_g, bb1_be, bb1_m, bb1_v, base0);
  k_backbone_topic<<<dim3(1, Cn, Bn), Mn, 0, stream>>>(
      topic, bbt_w, bbt_b, bbt_g, bbt_be, bbt_m, bbt_v, tf);

  // ---- two EgoPartiteGNeXtC blocks ----
  for (int bi = 0; bi < 2; ++bi) {
    const float* xin = bi ? base1 : base0;
    float* outp = bi ? (float*)d_out : base1;
    const float* identity = bi ? snip : nullptr;

    k_pack_x<<<dim3(Tn / 32, Cn / 32, Bn), dim3(32, 8), 0, stream>>>(xin, xh);
    k_gemm_t1<<<dim3(Wn / 16, Tn / 16, Bn), 32, 0, stream>>>(t1wh[bi], xh, g[bi].t1b, t1o);
    k_t2<<<dim3(Tn / 256, Wn, Bn), 256, 0, stream>>>(t1o, g[bi].t2w, g[bi].t2b, t2h);
    k_dist_topk<<<dim3(Tn, 1, Bn), Mn, 0, stream>>>(xin, tf, idxb);
    k_pmat<<<dim3(1, Wn, Bn), Mn, 0, stream>>>(g[bi].s1w, tf, pbuf);
    k_gemm_q<<<dim3(Wn / 16, Tn / 16, Bn), 32, 0, stream>>>(s1dh[bi], xh, qbuf);
    k_s1s2<<<dim3(Kk, Tn, Bn), Wn, 0, stream>>>(
        pbuf, qbuf, idxb, g[bi].s1b, g[bi].s2w, g[bi].s2b, s2h);
    k_final<<<dim3(Cn / 16, Tn / 16, Bn), 32, 0, stream>>>(
        t3wh[bi], s3wh[bi], t2h, s2h, xin, g[bi].t3b, g[bi].s3b, identity, outp);
  }
}
